// Gtu2d_35433480192024
// MI455X (gfx1250) — matrix operand order repaired and checked by reading, not compiled, not run on hardware
//
#include <hip/hip_runtime.h>
#include <hip/hip_bf16.h>
#include <math.h>

typedef _Float16 h16;
typedef __attribute__((ext_vector_type(16))) _Float16 v16h;
typedef __attribute__((ext_vector_type(8)))  float    v8f;

#define B_   8
#define N_   4096
#define DM   512
#define D1   1024
#define NH_  8
#define HD_  128

__device__ __forceinline__ int lane_id() { return threadIdx.x & 31; }

__device__ __forceinline__ v8f wmma_f16(v16h a, v16h b, v8f c) {
  return __builtin_amdgcn_wmma_f32_16x16x32_f16(false, a, false, b, (short)0, c,
                                                false, false);
}

__device__ __forceinline__ float silu_f(float x) { return x / (1.f + __expf(-x)); }

// ---- CDNA5 async global->LDS copy (ASYNCcnt-tracked, §15.18.3) ------------
__device__ __forceinline__ void async_g2l_b128(void* lds_ptr, const void* sbase,
                                               int voff_bytes) {
  unsigned lds_addr = (unsigned)(unsigned long long)lds_ptr;  // low 32 = LDS offset
  asm volatile("global_load_async_to_lds_b128 %0, %1, %2"
               :
               : "v"(lds_addr), "v"(voff_bytes), "s"(sbase)
               : "memory");
}
__device__ __forceinline__ void wait_async0() {
  asm volatile("s_wait_asynccnt 0x0" ::: "memory");
}
__device__ __forceinline__ void wait_async2() {  // allow 2 newest in flight
  asm volatile("s_wait_asynccnt 0x2" ::: "memory");
}

// ---- WMMA fragment loaders (ISA 7.12.2 layouts, wave32) -------------------
__device__ __forceinline__ v16h load_a_f16(const h16* s, int ld, int m0, int k0) {
  int l = lane_id();
  int row = m0 + (l & 15);
  int hf = (l >> 4) * 8;
  const h16* p = s + row * ld + k0 + hf;
  v16h a;
#pragma unroll
  for (int r = 0; r < 4; ++r) { a[2 * r] = p[2 * r]; a[2 * r + 1] = p[2 * r + 1]; }
  const h16* q = p + 16;
#pragma unroll
  for (int r = 0; r < 4; ++r) { a[8 + 2 * r] = q[2 * r]; a[9 + 2 * r] = q[2 * r + 1]; }
  return a;
}

// A-fragment from K-major (transposed) storage: A[m][k] = s[k*ld + m]
__device__ __forceinline__ v16h load_aT_f16(const h16* s, int ld, int m0, int k0) {
  int l = lane_id();
  int row = m0 + (l & 15);
  int hf = (l >> 4) * 8;
  v16h a;
#pragma unroll
  for (int r = 0; r < 4; ++r) {
    int k = k0 + hf + 2 * r;
    a[2 * r]     = s[k * ld + row];
    a[2 * r + 1] = s[(k + 1) * ld + row];
  }
#pragma unroll
  for (int r = 0; r < 4; ++r) {
    int k = k0 + 16 + hf + 2 * r;
    a[8 + 2 * r] = s[k * ld + row];
    a[9 + 2 * r] = s[(k + 1) * ld + row];
  }
  return a;
}

__device__ __forceinline__ v16h load_b_f16(const h16* s, int ld, int k0, int n0) {
  int l = lane_id();
  int col = n0 + (l & 15);
  int kb = k0 + ((l >> 4) << 3);
  v16h b;
#pragma unroll
  for (int r = 0; r < 8; ++r) {
    int k = kb + 2 * (r & 3) + ((r >> 2) << 4);
    b[2 * r]     = s[k * ld + col];
    b[2 * r + 1] = s[(k + 1) * ld + col];
  }
  return b;
}

__device__ __forceinline__ v16h load_b_f32g(const float* s, int ld, int k0, int n0) {
  int l = lane_id();
  int col = n0 + (l & 15);
  int kb = k0 + ((l >> 4) << 3);
  v16h b;
#pragma unroll
  for (int r = 0; r < 8; ++r) {
    int k = kb + 2 * (r & 3) + ((r >> 2) << 4);
    b[2 * r]     = (h16)s[k * ld + col];
    b[2 * r + 1] = (h16)s[(k + 1) * ld + col];
  }
  return b;
}

// Toeplitz A-fragment: A[row][k] = t[(row-k)&127]
__device__ __forceinline__ v16h load_a_toep(const float* tab, int m0, int k0) {
  int l = lane_id();
  int row = m0 + (l & 15);
  int hf = (l >> 4) * 8;
  v16h a;
#pragma unroll
  for (int r = 0; r < 4; ++r) {
    int k = k0 + hf + 2 * r;
    a[2 * r]     = (h16)tab[(row - k) & 127];
    a[2 * r + 1] = (h16)tab[(row - k - 1) & 127];
  }
#pragma unroll
  for (int r = 0; r < 4; ++r) {
    int k = k0 + 16 + hf + 2 * r;
    a[8 + 2 * r] = (h16)tab[(row - k) & 127];
    a[9 + 2 * r] = (h16)tab[(row - k - 1) & 127];
  }
  return a;
}

// Toeplitz-transpose B-fragment: B[k][col] = t[(col-k)&127]
__device__ __forceinline__ v16h load_bT_toep(const float* tab, int k0, int n0) {
  int l = lane_id();
  int col = n0 + (l & 15);
  int kb = k0 + ((l >> 4) << 3);
  v16h b;
#pragma unroll
  for (int r = 0; r < 8; ++r) {
    int k = kb + 2 * (r & 3) + ((r >> 2) << 4);
    b[2 * r]     = (h16)tab[(col - k) & 127];
    b[2 * r + 1] = (h16)tab[(col - (k + 1)) & 127];
  }
  return b;
}

// ---- Kernel 1: RPE MLP -> Toeplitz coefficient table [8][128][128] --------
__global__ __launch_bounds__(256) void rpe_kernel(
    const float* __restrict__ w_in, const float* __restrict__ b_in,
    const float* __restrict__ wh, const float* __restrict__ bh,
    const float* __restrict__ w_out, const float* __restrict__ b_out,
    float* __restrict__ tab) {
  __shared__ h16 act[32 * 520];
  const int tid = threadIdx.x;
  const int wid = tid >> 5;
  const int m0 = blockIdx.x * 32;
  for (int i = tid; i < 32 * 512; i += 256) {
    int r = i >> 9, c = i & 511;
    int mg = m0 + r;
    float idxv = (mg < 64) ? (float)mg : ((mg == 127) ? 0.f : (float)(mg - 127));
    float v = idxv * w_in[c] + b_in[c];
    act[r * 520 + c] = (h16)fmaxf(v, 0.f);
  }
  __syncthreads();
  const int l = lane_id();
  const int cl = l & 15;
  const int rb = (l >> 4) * 8;
  for (int L = 0; L < 3; ++L) {
    const float* W = wh + (size_t)L * 512 * 512;
    const float* bb = bh + L * 512;
    v8f acc[8];
#pragma unroll
    for (int t = 0; t < 8; ++t) {
      int T = wid * 8 + t;
      int nt = T & 31, mt = T >> 5;
      v8f c = {};
#pragma unroll 1
      for (int kc = 0; kc < 16; ++kc) {
        v16h a = load_a_f16(act, 520, mt * 16, kc * 32);
        v16h b = load_b_f32g(W, 512, kc * 32, nt * 16);
        c = wmma_f16(a, b, c);
      }
      acc[t] = c;
    }
    __syncthreads();
#pragma unroll
    for (int t = 0; t < 8; ++t) {
      int T = wid * 8 + t;
      int nt = T & 31, mt = T >> 5;
#pragma unroll
      for (int r = 0; r < 8; ++r) {
        float v = acc[t][r] + bb[nt * 16 + cl];
        act[(mt * 16 + rb + r) * 520 + nt * 16 + cl] = (h16)fmaxf(v, 0.f);
      }
    }
    __syncthreads();
  }
#pragma unroll 1
  for (int t = 0; t < 16; ++t) {
    int T = wid * 16 + t;
    int nt = T & 63, mt = T >> 6;
    v8f c = {};
#pragma unroll 1
    for (int kc = 0; kc < 16; ++kc) {
      v16h a = load_a_f16(act, 520, mt * 16, kc * 32);
      v16h b = load_b_f32g(w_out, 1024, kc * 32, nt * 16);
      c = wmma_f16(a, b, c);
    }
#pragma unroll
    for (int r = 0; r < 8; ++r) {
      int mg = m0 + mt * 16 + rb + r;
      int cg = nt * 16 + cl;
      int hh = cg >> 7, dd = cg & 127;
      int pos = (mg < 64) ? mg : ((mg == 127) ? 64 : mg + 1);
      float v = (mg == 127) ? 0.f : (c[r] + b_out[cg]);
      tab[((size_t)hh * 128 + pos) * 128 + dd] = v;
    }
  }
}

// ---- Kernel 2: projection silu(x@W+b) -> permuted [b,slice,n] f16 ---------
// Double-buffered LDS staging; 128x128 tile/block; 8 waves in 2x4.
__device__ __forceinline__ void stage_proj(const float* X, const float* Wt,
                                           h16* Asb, h16* Bsb, int m0g, int n0g,
                                           int k0, int tid) {
#pragma unroll
  for (int it = 0; it < 4; ++it) {  // A tile 128x32 f32->f16
    int fi = it * 256 + tid;
    int row = fi >> 3, c4 = (fi & 7) * 4;
    float4 v = *(const float4*)(X + (size_t)(m0g + row) * DM + k0 + c4);
    Asb[row * 40 + c4 + 0] = (h16)v.x; Asb[row * 40 + c4 + 1] = (h16)v.y;
    Asb[row * 40 + c4 + 2] = (h16)v.z; Asb[row * 40 + c4 + 3] = (h16)v.w;
  }
#pragma unroll
  for (int it = 0; it < 4; ++it) {  // B tile 32x128 f32->f16
    int fi = it * 256 + tid;
    int row = fi >> 5, c4 = (fi & 31) * 4;
    float4 v = *(const float4*)(Wt + (size_t)(k0 + row) * D1 + n0g + c4);
    Bsb[row * 136 + c4 + 0] = (h16)v.x; Bsb[row * 136 + c4 + 1] = (h16)v.y;
    Bsb[row * 136 + c4 + 2] = (h16)v.z; Bsb[row * 136 + c4 + 3] = (h16)v.w;
  }
}

__global__ __launch_bounds__(256) void proj_kernel(
    const float* __restrict__ X, const float* __restrict__ Wt,
    const float* __restrict__ bias, h16* __restrict__ outp) {
  __shared__ h16 As[2][128 * 40];
  __shared__ h16 Bs[2][32 * 136];
  const int tid = threadIdx.x;
  const int wid = tid >> 5;
  const int wm = wid & 1, wn = wid >> 1;
  const int m0g = blockIdx.x * 128, n0g = blockIdx.y * 128;
  v8f acc[8];
#pragma unroll
  for (int i = 0; i < 8; ++i) acc[i] = (v8f){};
  stage_proj(X, Wt, As[0], Bs[0], m0g, n0g, 0, tid);
#pragma unroll 1
  for (int kc = 0; kc < 16; ++kc) {
    int cur = kc & 1;
    if (kc + 1 < 16)  // stage next chunk into the other buffer
      stage_proj(X, Wt, As[cur ^ 1], Bs[cur ^ 1], m0g, n0g, (kc + 1) * 32, tid);
    __syncthreads();
    v16h b0 = load_b_f16(Bs[cur], 136, 0, wn * 32);
    v16h b1 = load_b_f16(Bs[cur], 136, 0, wn * 32 + 16);
#pragma unroll
    for (int mt = 0; mt < 4; ++mt) {
      v16h a = load_a_f16(As[cur], 40, wm * 64 + mt * 16, 0);
      acc[mt * 2 + 0] = wmma_f16(a, b0, acc[mt * 2 + 0]);
      acc[mt * 2 + 1] = wmma_f16(a, b1, acc[mt * 2 + 1]);
    }
    __syncthreads();
  }
  const int l = lane_id();
  const int cl = l & 15;
  const int rb = (l >> 4) * 8;
#pragma unroll
  for (int mt = 0; mt < 4; ++mt)
#pragma unroll
    for (int nt = 0; nt < 2; ++nt)
#pragma unroll
      for (int r = 0; r < 8; ++r) {
        int row = m0g + wm * 64 + mt * 16 + rb + r;
        int col = n0g + wn * 32 + nt * 16 + cl;
        float v = silu_f(acc[mt * 2 + nt][r] + bias[col]);
        int b = row >> 12, rem = row & 4095;
        outp[((size_t)b * D1 + col) * (size_t)N_ + rem] = (h16)v;  // slice-major
      }
}

// ---- Kernel 3: 2-D Toeplitz mix, fused gate -------------------------------
// out = 2 * T2(H-mix) @ X @ T1(W-mix)^T  (o1 == o2 by associativity)
// 2 waves/block, one (b,h,d) slice per wave; X async-staged into LDS;
// u read and g written slice-contiguously.
__global__ __launch_bounds__(64) void toeplitz_kernel(
    const float* __restrict__ tab1, const float* __restrict__ tab2,
    const h16* __restrict__ vperm, const h16* __restrict__ uperm,
    h16* __restrict__ gperm) {
  __shared__ float t1s[2][128];
  __shared__ float t2s[2][128];
  __shared__ h16 xs[2][64 * 72];
  __shared__ h16 zb[2][64 * 72];
  const int tid = threadIdx.x;
  const int wid = tid >> 5;
  const int d0 = blockIdx.x * 2;
  const int h = blockIdx.y;
  const int b = blockIdx.z;
  for (int i = tid; i < 256; i += 64) {
    int dw = i >> 7, p = i & 127;
    t1s[dw][p] = tab1[((size_t)h * 128 + p) * 128 + d0 + dw];
    t2s[dw][p] = tab2[((size_t)h * 128 + p) * 128 + d0 + dw];
  }
  const size_t slice = ((size_t)b * NH_ + h) * HD_ + d0 + wid;
  const h16* Xp = vperm + slice * (size_t)N_;
  const h16* Up = uperm + slice * (size_t)N_;
  h16* Gp = gperm + slice * (size_t)N_;
  const int l = lane_id();
  {  // async-stage this wave's 64x64 f16 slice: 16 x B128 (4 rows of 128B)
    int row_in = l >> 3;
    int colh = (l & 7) * 8;
#pragma unroll
    for (int i = 0; i < 16; ++i) {
      int row = i * 4 + row_in;
      async_g2l_b128(&xs[wid][row * 72 + colh], Xp, (row * 64 + colh) * 2);
    }
    wait_async0();
  }
  __syncthreads();
  const int cl = l & 15;
  const int rb = (l >> 4) * 8;
  h16* z = zb[wid];
  // Z = T2 @ X
#pragma unroll 1
  for (int nt = 0; nt < 4; ++nt) {
    v16h bx0 = load_b_f16(xs[wid], 72, 0, nt * 16);
    v16h bx1 = load_b_f16(xs[wid], 72, 32, nt * 16);
#pragma unroll
    for (int mt = 0; mt < 4; ++mt) {
      v8f c = {};
      c = wmma_f16(load_a_toep(t2s[wid], mt * 16, 0), bx0, c);
      c = wmma_f16(load_a_toep(t2s[wid], mt * 16, 32), bx1, c);
#pragma unroll
      for (int r = 0; r < 8; ++r)
        z[(mt * 16 + rb + r) * 72 + nt * 16 + cl] = (h16)c[r];
    }
  }
  // Y = Z @ T1^T ; epilogue: g = u * 2*Y, contiguous in slice layout
#pragma unroll 1
  for (int mt = 0; mt < 4; ++mt)
#pragma unroll 1
    for (int nt = 0; nt < 4; ++nt) {
      v8f c = {};
      c = wmma_f16(load_a_f16(z, 72, mt * 16, 0),
                   load_bT_toep(t1s[wid], 0, nt * 16), c);
      c = wmma_f16(load_a_f16(z, 72, mt * 16, 32),
                   load_bT_toep(t1s[wid], 32, nt * 16), c);
#pragma unroll
      for (int r = 0; r < 8; ++r) {
        int idx = (mt * 16 + rb + r) * 64 + nt * 16 + cl;
        float uv = (float)Up[idx];
        Gp[idx] = (h16)(2.f * c[r] * uv);
      }
    }
}

// ---- Kernel 4: output projection out = g @ wo + bo ------------------------
// A is gperm, K-major by construction: async-staged [k][n] rows; pipelined.
__device__ __forceinline__ void issue_A_async(const h16* Gbase, h16* dst, int k0,
                                              int wid, int l) {
  int kr0 = wid * 4 + (l >> 4);   // rows kr0, kr0+2 (16 lanes x 16B per row)
  int colh = (l & 15) * 8;
#pragma unroll
  for (int j = 0; j < 2; ++j) {
    int kr = kr0 + j * 2;
    async_g2l_b128(dst + kr * 136 + colh, Gbase,
                   (int)(((k0 + kr) * N_ + colh) * 2));
  }
}

__device__ __forceinline__ void stage_B_wo(const float* Wo, h16* Bsb, int n0g,
                                           int k0, int tid) {
#pragma unroll
  for (int it = 0; it < 4; ++it) {  // 32x128 f32->f16, ld=512
    int fi = it * 256 + tid;
    int row = fi >> 5, c4 = (fi & 31) * 4;
    float4 v = *(const float4*)(Wo + (size_t)(k0 + row) * DM + n0g + c4);
    Bsb[row * 136 + c4 + 0] = (h16)v.x; Bsb[row * 136 + c4 + 1] = (h16)v.y;
    Bsb[row * 136 + c4 + 2] = (h16)v.z; Bsb[row * 136 + c4 + 3] = (h16)v.w;
  }
}

__global__ __launch_bounds__(256) void outproj_kernel(
    const h16* __restrict__ G, const float* __restrict__ Wo,
    const float* __restrict__ bo, float* __restrict__ out) {
  __shared__ h16 Ats[2][32 * 136];  // [k][n] (K-major) A tiles
  __shared__ h16 Bs[2][32 * 136];
  const int tid = threadIdx.x;
  const int wid = tid >> 5;
  const int wm = wid & 1, wn = wid >> 1;
  const int m0g = blockIdx.x * 128, n0g = blockIdx.y * 128;
  const int l = lane_id();
  const int bb = m0g >> 12, nbase = m0g & 4095;
  const h16* Gbase = G + ((size_t)bb * D1) * (size_t)N_ + nbase;  // uniform base
  v8f acc[8];
#pragma unroll
  for (int i = 0; i < 8; ++i) acc[i] = (v8f){};
  issue_A_async(Gbase, Ats[0], 0, wid, l);
  stage_B_wo(Wo, Bs[0], n0g, 0, tid);
#pragma unroll 1
  for (int kc = 0; kc < 32; ++kc) {
    int cur = kc & 1;
    if (kc + 1 < 32) {  // pipeline: copies for next chunk overlap this compute
      issue_A_async(Gbase, Ats[cur ^ 1], (kc + 1) * 32, wid, l);
      stage_B_wo(Wo, Bs[cur ^ 1], n0g, (kc + 1) * 32, tid);
      wait_async2();  // drain older buffer's copies; keep 2 newest in flight
    } else {
      wait_async0();
    }
    __syncthreads();
    v16h b0 = load_b_f16(Bs[cur], 136, 0, wn * 32);
    v16h b1 = load_b_f16(Bs[cur], 136, 0, wn * 32 + 16);
#pragma unroll
    for (int mt = 0; mt < 4; ++mt) {
      v16h a = load_aT_f16(Ats[cur], 136, wm * 64 + mt * 16, 0);
      acc[mt * 2 + 0] = wmma_f16(a, b0, acc[mt * 2 + 0]);
      acc[mt * 2 + 1] = wmma_f16(a, b1, acc[mt * 2 + 1]);
    }
    __syncthreads();
  }
  const int cl = l & 15;
  const int rb = (l >> 4) * 8;
#pragma unroll
  for (int mt = 0; mt < 4; ++mt)
#pragma unroll
    for (int nt = 0; nt < 2; ++nt)
#pragma unroll
      for (int r = 0; r < 8; ++r) {
        int row = m0g + wm * 64 + mt * 16 + rb + r;
        int col = n0g + wn * 32 + nt * 16 + cl;
        out[(size_t)row * DM + col] = acc[mt * 2 + nt][r] + bo[col];
      }
}

// ---- Host launcher --------------------------------------------------------
extern "C" void kernel_launch(void* const* d_in, const int* in_sizes, int n_in,
                              void* d_out, int out_size, void* d_ws, size_t ws_size,
                              hipStream_t stream) {
  const float* x  = (const float*)d_in[0];
  const float* wu = (const float*)d_in[1];
  const float* bu = (const float*)d_in[2];
  const float* wv = (const float*)d_in[3];
  const float* bv = (const float*)d_in[4];
  const float* wo = (const float*)d_in[5];
  const float* bo = (const float*)d_in[6];
  const float* p1_w_in  = (const float*)d_in[7];
  const float* p1_b_in  = (const float*)d_in[8];
  const float* p1_wh    = (const float*)d_in[9];
  const float* p1_bh    = (const float*)d_in[10];
  const float* p1_w_out = (const float*)d_in[11];
  const float* p1_b_out = (const float*)d_in[12];
  const float* p2_w_in  = (const float*)d_in[13];
  const float* p2_b_in  = (const float*)d_in[14];
  const float* p2_wh    = (const float*)d_in[15];
  const float* p2_bh    = (const float*)d_in[16];
  const float* p2_w_out = (const float*)d_in[17];
  const float* p2_b_out = (const float*)d_in[18];

  char* ws = (char*)d_ws;
  const size_t SLICES = (size_t)B_ * D1;                    // 8192 slices of 4096
  const size_t OFF_TAB1 = 0;                                // 512 KB
  const size_t OFF_TAB2 = OFF_TAB1 + 524288;
  const size_t OFF_U    = OFF_TAB2 + 524288;                // f16 permuted, 64 MB
  const size_t OFF_V    = OFF_U + SLICES * N_ * 2;          // f16 permuted, 64 MB
  const size_t OFF_G    = OFF_V + SLICES * N_ * 2;          // f16 permuted, 64 MB
  float* tab1 = (float*)(ws + OFF_TAB1);
  float* tab2 = (float*)(ws + OFF_TAB2);
  h16* uperm  = (h16*)(ws + OFF_U);
  h16* vperm  = (h16*)(ws + OFF_V);
  h16* gperm  = (h16*)(ws + OFF_G);
  float* out  = (float*)d_out;

  rpe_kernel<<<dim3(4), 256, 0, stream>>>(p1_w_in, p1_b_in, p1_wh, p1_bh,
                                          p1_w_out, p1_b_out, tab1);
  rpe_kernel<<<dim3(4), 256, 0, stream>>>(p2_w_in, p2_b_in, p2_wh, p2_bh,
                                          p2_w_out, p2_b_out, tab2);
  proj_kernel<<<dim3(256, 8), 256, 0, stream>>>(x, wu, bu, uperm);
  proj_kernel<<<dim3(256, 8), 256, 0, stream>>>(x, wv, bv, vperm);
  toeplitz_kernel<<<dim3(64, 8, 8), 64, 0, stream>>>(tab1, tab2, vperm, uperm, gperm);
  outproj_kernel<<<dim3(256, 4), 256, 0, stream>>>(gperm, wo, bo, out);
}
